// AdaptiveGraphLayer_2748779070194
// MI455X (gfx1250) — matrix operand order repaired and checked by reading, not compiled, not run on hardware
//
#include <hip/hip_runtime.h>
#include <hip/hip_bf16.h>
#include <math.h>

// ---------------- problem dims ----------------
#define N_NODES 4096
#define F_IN    256
#define OUT_F   256
#define NHEAD   4
#define HDIM    64

// ---------------- ws layout (half-element offsets) ----------------
// x_bf   : [4096][256] bf16            @ 0         (2 MB)
// W_bf   : 3x[256][256] bf16           @ 1048576   (384 KB)
// Q_bf   : [4096][256] bf16            @ 1245184   (2 MB)
// K_bf   : [4096][256] bf16            @ 2293760   (2 MB)
// Vt_bf  : [256][4096] bf16            @ 3342336   (2 MB)
// adj    : [4096][128] u32 bitmask     @ byte 8781824 (2 MB)
#define WS_XBF   0u
#define WS_WBF   1048576u
#define WS_QBF   1245184u
#define WS_KBF   2293760u
#define WS_VTBF  3342336u
#define WS_ADJ_BYTES 8781824u

typedef float  v8f   __attribute__((ext_vector_type(8)));
typedef __bf16 v16bf __attribute__((ext_vector_type(16)));
typedef unsigned int uv4 __attribute__((ext_vector_type(4)));

union FragB { v16bf v; uv4 q[2]; unsigned short s[16]; };
union Acc   { v8f  v; float f[8]; };

__device__ __forceinline__ unsigned short f2bf(float f) {
  unsigned u = __float_as_uint(f);
  u += 0x7FFFu + ((u >> 16) & 1u);          // round-to-nearest-even
  return (unsigned short)(u >> 16);
}

__device__ __forceinline__ v8f wmma_bf16(const FragB& a, const FragB& b, v8f c) {
  return __builtin_amdgcn_wmma_f32_16x16x32_bf16(false, a.v, false, b.v,
                                                 (short)0, c, false, false);
}

// ---------------- 1) convert x,W -> bf16 ; zero adjacency ----------------
__global__ void k_convert(const float* __restrict__ x,
                          const float* __restrict__ wq,
                          const float* __restrict__ wk,
                          const float* __restrict__ wv,
                          unsigned short* __restrict__ ws16,
                          unsigned* __restrict__ adj) {
  unsigned idx = blockIdx.x * 256u + threadIdx.x;
  if (idx < 1048576u) {
    ws16[WS_XBF + idx] = f2bf(x[idx]);
  } else if (idx < 1245184u) {
    unsigned w = idx - 1048576u;
    const float* src = (w < 65536u) ? wq : (w < 131072u) ? wk : wv;
    ws16[WS_WBF + w] = f2bf(src[w & 65535u]);
  } else if (idx < 1769472u) {
    adj[idx - 1245184u] = 0u;
  }
}

// ---------------- 2) scatter edges into bitmask ----------------
__global__ void k_adj(const int* __restrict__ ei, int E, unsigned* __restrict__ adj) {
  int i = blockIdx.x * 256 + threadIdx.x;
  if (i < E) {
    int n = ei[i];
    int m = ei[E + i];
    atomicOr(&adj[n * 128 + (m >> 5)], 1u << (m & 31));
  }
}

// ---------------- 3) QKV GEMM: [4096x256] = x_bf16 @ W^T + b ----------------
// one wave -> one 16x16 output tile, K-loop of 8 WMMAs (K=32 each)
__global__ void k_qkv(unsigned short* __restrict__ ws16,
                      const float* __restrict__ bq,
                      const float* __restrict__ bk,
                      const float* __restrict__ bv) {
  const unsigned short* xbf = ws16 + WS_XBF;
  const unsigned short* wbf = ws16 + WS_WBF;
  unsigned short* Qbf  = ws16 + WS_QBF;
  unsigned short* Kbf  = ws16 + WS_KBF;
  unsigned short* Vtbf = ws16 + WS_VTBF;

  int lane = threadIdx.x & 31;
  int wid  = threadIdx.x >> 5;
  int id   = blockIdx.x * 4 + wid;      // 0..12287
  int mat  = id >> 12;                  // 0=Q,1=K,2=V
  int t    = id & 4095;
  int rowt = t >> 4;
  int colt = t & 15;

  const unsigned short* W = wbf + mat * 65536;
  int rA    = rowt * 16 + (lane & 15);
  int koffA = (lane & 16) ? 16 : 0;
  int o     = colt * 16 + (lane & 15);
  int koffB = (lane & 16) ? 16 : 0;

  Acc c;
#pragma unroll
  for (int j = 0; j < 8; ++j) c.f[j] = 0.0f;

#pragma unroll
  for (int k0 = 0; k0 < 256; k0 += 32) {
    FragB a, b;
    const unsigned short* ap = xbf + rA * 256 + k0 + koffA;
    a.q[0] = *(const uv4*)(ap);
    a.q[1] = *(const uv4*)(ap + 8);
    const unsigned short* bp = W + o * 256 + k0 + koffB;
    b.q[0] = *(const uv4*)(bp);
    b.q[1] = *(const uv4*)(bp + 8);
    c.v = wmma_bf16(a, b, c.v);
  }

  const float* bias = (mat == 0) ? bq : (mat == 1) ? bk : bv;
  float bval = bias[o];
  int hi8 = (lane & 16) ? 8 : 0;
#pragma unroll
  for (int j = 0; j < 8; ++j) {
    int row = rowt * 16 + j + hi8;
    float val = c.f[j] + bval;
    if (mat == 0)      Qbf[row * 256 + o] = f2bf(val);
    else if (mat == 1) Kbf[row * 256 + o] = f2bf(val);
    else               Vtbf[o * 4096 + row] = f2bf(val);   // transposed V
  }
}

// ---------------- 4) attention: per-wave 16-row strip of one head ----------------
__global__ void k_attn(unsigned short* __restrict__ ws16,
                       const unsigned* __restrict__ adj,
                       float* __restrict__ attn,
                       float* __restrict__ out0) {
  __shared__ unsigned        adjS[4][16 * 128];    // 32 KB
  __shared__ unsigned short  estage[4][16 * 32];   // 4 KB
  __shared__ float           invS[4][16];
  __shared__ float           rmS[4][16];

  const unsigned short* Qbf  = ws16 + WS_QBF;
  const unsigned short* Kbf  = ws16 + WS_KBF;
  const unsigned short* Vtbf = ws16 + WS_VTBF;

  int lane = threadIdx.x & 31;
  int wid  = threadIdx.x >> 5;
  int gid  = blockIdx.x * 4 + wid;      // 0..1023
  int head = gid >> 8;
  int strip = gid & 255;
  int r0 = strip * 16;
  int hi8 = (lane & 16) ? 8 : 0;

  // stage this strip's adjacency rows into LDS (16 rows x 128 words)
  for (int i = lane; i < 2048; i += 32)
    adjS[wid][i] = adj[r0 * 128 + i];

  // preload Q A-fragments for the whole strip (K = 0..63 in two frags)
  FragB qf[2];
  {
    int rqA = r0 + (lane & 15);
    int koffA = (lane & 16) ? 16 : 0;
#pragma unroll
    for (int f = 0; f < 2; ++f) {
      const unsigned short* qp = Qbf + rqA * 256 + head * 64 + f * 32 + koffA;
      qf[f].q[0] = *(const uv4*)(qp);
      qf[f].q[1] = *(const uv4*)(qp + 8);
    }
  }

  float* arow = attn + ((size_t)(head * 4096 + r0)) * 4096;

  // ---- phase 1: scores = QK^T/8, mask, write raw, track rowmax ----
  float rowm[8];
#pragma unroll
  for (int j = 0; j < 8; ++j) rowm[j] = -3.0e38f;

  int koffB = (lane & 16) ? 16 : 0;
  for (int ct = 0; ct < 256; ++ct) {
    int c0 = ct * 16;
    int colK = c0 + (lane & 15);
    const unsigned short* kb = Kbf + colK * 256 + head * 64 + koffB;
    FragB b0, b1;
    b0.q[0] = *(const uv4*)(kb);
    b0.q[1] = *(const uv4*)(kb + 8);
    b1.q[0] = *(const uv4*)(kb + 32);
    b1.q[1] = *(const uv4*)(kb + 40);
    if (ct < 255) __builtin_prefetch(kb + 16 * 256, 0, 0);  // next K tile

    Acc c;
#pragma unroll
    for (int j = 0; j < 8; ++j) c.f[j] = 0.0f;
    c.v = wmma_bf16(qf[0], b0, c.v);
    c.v = wmma_bf16(qf[1], b1, c.v);

    int wq  = c0 >> 5;
    int bsh = (c0 & 16) + (lane & 15);
    float* sp = arow + c0 + (lane & 15);
#pragma unroll
    for (int j = 0; j < 8; ++j) {
      int row = j + hi8;
      unsigned w = adjS[wid][row * 128 + wq];
      float s = ((w >> bsh) & 1u) ? c.f[j] * 0.125f : -3.0e38f;
      rowm[j] = fmaxf(rowm[j], s);
      sp[(size_t)row * 4096] = s;
    }
  }
  // reduce rowmax across the 16 lanes holding each row
#pragma unroll
  for (int j = 0; j < 8; ++j) {
#pragma unroll
    for (int m = 1; m <= 8; m <<= 1)
      rowm[j] = fmaxf(rowm[j], __shfl_xor(rowm[j], m, 32));
  }
  if ((lane & 15) == 0) {
#pragma unroll
    for (int j = 0; j < 8; ++j) rmS[wid][j + hi8] = rowm[j];
  }

  // ---- phase 2: e = exp(s-max); rowsum; O += e @ V (WMMA); no writeback ----
  float rs[8];
  Acc oacc[4];
#pragma unroll
  for (int j = 0; j < 8; ++j) rs[j] = 0.0f;
#pragma unroll
  for (int t = 0; t < 4; ++t)
#pragma unroll
    for (int j = 0; j < 8; ++j) oacc[t].f[j] = 0.0f;

  unsigned short* es = &estage[wid][0];
  for (int mt = 0; mt < 128; ++mt) {
    int m0 = mt * 32;
#pragma unroll
    for (int j = 0; j < 8; ++j) {
      int row = j + hi8;
      size_t off = (size_t)row * 4096 + m0 + (lane & 15);
      float e0 = __expf(arow[off] - rowm[j]);
      float e1 = __expf(arow[off + 16] - rowm[j]);
      rs[j] += e0 + e1;
      es[row * 32 + (lane & 15)]      = f2bf(e0);
      es[row * 32 + 16 + (lane & 15)] = f2bf(e1);
    }
    // A fragment (16x32 bf16) from staged e
    FragB af;
    {
      const unsigned short* ep = es + (lane & 15) * 32 + ((lane & 16) ? 16 : 0);
      af.q[0] = *(const uv4*)(ep);
      af.q[1] = *(const uv4*)(ep + 8);
    }
#pragma unroll
    for (int t = 0; t < 4; ++t) {
      FragB bf_;
      const unsigned short* vb =
          Vtbf + (size_t)(head * 64 + t * 16 + (lane & 15)) * 4096 + m0 + koffB;
      bf_.q[0] = *(const uv4*)(vb);
      bf_.q[1] = *(const uv4*)(vb + 8);
      oacc[t].v = wmma_bf16(af, bf_, oacc[t].v);
    }
  }

  // reduce rowsums, write O (pre-LN) and strip inverse sums
  float invv[8];
#pragma unroll
  for (int j = 0; j < 8; ++j) {
#pragma unroll
    for (int m = 1; m <= 8; m <<= 1)
      rs[j] += __shfl_xor(rs[j], m, 32);
    invv[j] = 1.0f / rs[j];
  }
  if ((lane & 15) == 0) {
#pragma unroll
    for (int j = 0; j < 8; ++j) invS[wid][j + hi8] = invv[j];
  }
#pragma unroll
  for (int t = 0; t < 4; ++t) {
#pragma unroll
    for (int j = 0; j < 8; ++j) {
      int node = r0 + j + hi8;
      int feat = head * 64 + t * 16 + (lane & 15);
      out0[node * 256 + feat] = oacc[t].f[j] * invv[j];
    }
  }

  // ---- phase 3: raw scores -> normalized softmax, in place ----
  float4* ab4 = (float4*)arow;                // 16*4096 floats contiguous
  for (int i = lane; i < 16384; i += 32) {
    int row = i >> 10;
    float inv = invS[wid][row];
    float rm  = rmS[wid][row];
    float4 v = ab4[i];
    v.x = __expf(v.x - rm) * inv;
    v.y = __expf(v.y - rm) * inv;
    v.z = __expf(v.z - rm) * inv;
    v.w = __expf(v.w - rm) * inv;
    ab4[i] = v;
  }
}

// ---------------- 5) residual + LayerNorm (one wave per node) ----------------
__global__ void k_ln(float* __restrict__ out0,
                     const float* __restrict__ x,
                     const float* __restrict__ g,
                     const float* __restrict__ b) {
  int lane = threadIdx.x & 31;
  int wid  = threadIdx.x >> 5;
  int node = blockIdx.x * 8 + wid;
  int base = node * 256 + lane * 8;

  float4 a0 = *(float4*)(out0 + base);
  float4 a1 = *(float4*)(out0 + base + 4);
  float4 x0 = *(const float4*)(x + base);
  float4 x1 = *(const float4*)(x + base + 4);
  float v[8] = {a0.x + x0.x, a0.y + x0.y, a0.z + x0.z, a0.w + x0.w,
                a1.x + x1.x, a1.y + x1.y, a1.z + x1.z, a1.w + x1.w};

  float sum = 0.0f;
#pragma unroll
  for (int k = 0; k < 8; ++k) sum += v[k];
#pragma unroll
  for (int m = 1; m <= 16; m <<= 1) sum += __shfl_xor(sum, m, 32);
  float mu = sum * (1.0f / 256.0f);

  float s2 = 0.0f;
#pragma unroll
  for (int k = 0; k < 8; ++k) { float d = v[k] - mu; s2 += d * d; }
#pragma unroll
  for (int m = 1; m <= 16; m <<= 1) s2 += __shfl_xor(s2, m, 32);
  float r = rsqrtf(s2 * (1.0f / 256.0f) + 1e-5f);

  float y[8];
#pragma unroll
  for (int k = 0; k < 8; ++k) {
    int f = lane * 8 + k;
    y[k] = (v[k] - mu) * r * g[f] + b[f];
  }
  *(float4*)(out0 + base)     = make_float4(y[0], y[1], y[2], y[3]);
  *(float4*)(out0 + base + 4) = make_float4(y[4], y[5], y[6], y[7]);
}

// ---------------- host wrapper ----------------
extern "C" void kernel_launch(void* const* d_in, const int* in_sizes, int n_in,
                              void* d_out, int out_size, void* d_ws, size_t ws_size,
                              hipStream_t stream) {
  (void)n_in; (void)out_size; (void)ws_size;
  const float* x   = (const float*)d_in[0];
  const int*   ei  = (const int*)d_in[1];
  const float* Wq  = (const float*)d_in[2];
  const float* bq  = (const float*)d_in[3];
  const float* Wk  = (const float*)d_in[4];
  const float* bk  = (const float*)d_in[5];
  const float* Wv  = (const float*)d_in[6];
  const float* bv  = (const float*)d_in[7];
  const float* lng = (const float*)d_in[8];
  const float* lnb = (const float*)d_in[9];

  unsigned short* ws16 = (unsigned short*)d_ws;
  unsigned* adj = (unsigned*)((char*)d_ws + WS_ADJ_BYTES);
  float* out0 = (float*)d_out;
  float* attn = out0 + (size_t)N_NODES * OUT_F;
  int E = in_sizes[1] / 2;

  k_convert<<<6912, 256, 0, stream>>>(x, Wq, Wk, Wv, ws16, adj);
  k_adj<<<(E + 255) / 256, 256, 0, stream>>>(ei, E, adj);
  k_qkv<<<3072, 128, 0, stream>>>(ws16, bq, bk, bv);
  k_attn<<<256, 128, 0, stream>>>(ws16, adj, attn, out0);
  k_ln<<<512, 256, 0, stream>>>(out0, x, lng, lnb);
}
